// Attn_30683246362810
// MI455X (gfx1250) — compile-verified, matrix operand order checked by reading, not run on hardware
//
#include <hip/hip_runtime.h>
#include <hip/hip_bf16.h>

// ---------------- problem constants ----------------
#define DIMN     1024
#define Q_HEADS  16
#define KV_HEADS 4
#define HEAD_DIM 64
#define GQA_DIM  256            // HEAD_DIM * KV_HEADS
#define QKV_N    2560           // GQA_DIM*2 + DIMN*2
#define BATCH    8
#define SEQ      1024
#define TTOT     (BATCH * SEQ)  // 8192
#define EPSV     1e-6f

typedef __attribute__((ext_vector_type(16))) _Float16 v16h;
typedef __attribute__((ext_vector_type(8)))  _Float16 v8h;
typedef __attribute__((ext_vector_type(8)))  float    v8f;

// 16x16x32 f16 WMMA, f32 accumulate
__device__ __forceinline__ v8f wmma_f16(v16h a, v16h b, v8f c) {
  return __builtin_amdgcn_wmma_f32_16x16x32_f16(
      /*neg_a=*/false, a, /*neg_b=*/false, b,
      /*c_mod=*/(short)0, c, /*reuse_a=*/false, /*reuse_b=*/false);
}

// Async 16B copy global -> LDS (CDNA5 GLOBAL_LOAD_ASYNC_TO_LDS_B128, ASYNCcnt).
// Generic LDS pointers carry the LDS byte offset in their low 32 bits
// (ISA 10.2 aperture mapping), which is exactly the VDST operand.
__device__ __forceinline__ void async_copy_b128(const void* gptr, void* lptr) {
  unsigned          lds = (unsigned)(size_t)lptr;
  unsigned long long ga = (unsigned long long)(size_t)gptr;
  asm volatile("global_load_async_to_lds_b128 %0, %1, off"
               :: "v"(lds), "v"(ga) : "memory");
}
__device__ __forceinline__ void wait_async() {
  asm volatile("s_wait_asynccnt 0" ::: "memory");
}

// Load a 16x32 f16 A/B fragment from a row-major [16 x >=32] tile.
// ISA 7.12.2 layout: lane L holds row (L&15); elems 0..7 = K[ko..ko+7],
// elems 8..15 = K[16+ko..23+ko], ko = (L>>4)*8.  Two aligned 16B loads.
__device__ __forceinline__ v16h load_frag(const _Float16* base, int stride, int lane) {
  int r  = lane & 15;
  int ko = (lane >> 4) << 3;
  const _Float16* p = base + r * stride + ko;
  union { v16h v; v8h h[2]; } u;
  u.h[0] = *(const v8h*)(p);
  u.h[1] = *(const v8h*)(p + 16);
  return u.v;
}

// ---------------- kernel 1: RMSNorm + f16 cast ----------------
__global__ __launch_bounds__(256) void k_rmsnorm(const float* __restrict__ x,
                                                 const float* __restrict__ g,
                                                 _Float16* __restrict__ h) {
  int row = blockIdx.x;
  const float* xr = x + (size_t)row * DIMN;
  float ss = 0.f;
  for (int c = threadIdx.x; c < DIMN; c += 256) { float v = xr[c]; ss += v * v; }
  for (int m = 16; m >= 1; m >>= 1) ss += __shfl_xor(ss, m, 32);
  __shared__ float red[8];
  if ((threadIdx.x & 31) == 0) red[threadIdx.x >> 5] = ss;
  __syncthreads();
  float tot = 0.f;
  for (int w = 0; w < 8; ++w) tot += red[w];
  float rs = rsqrtf(tot / (float)DIMN + EPSV);
  _Float16* hr = h + (size_t)row * DIMN;
  for (int c = threadIdx.x; c < DIMN; c += 256)
    hr[c] = (_Float16)(xr[c] * rs * g[c]);
}

// ---------------- kernel 2: f32 -> f16 cast ----------------
__global__ __launch_bounds__(256) void k_cvt16(const float* __restrict__ src,
                                               _Float16* __restrict__ dst, int n) {
  int i = blockIdx.x * 256 + threadIdx.x;
  if (i < n) dst[i] = (_Float16)src[i];
}

// ---------------- kernel 3/6: WMMA GEMM  C = A @ B^T ----------------
// A: [M x K] f16 row-major, B: [N x K] f16 row-major. Block tile 128x64,
// 8 waves in 4(M) x 2(N); each wave 32x32 = 2x2 WMMA tiles, K-step 32.
// Double-buffered LDS staged by GLOBAL_LOAD_ASYNC_TO_LDS_B128: next slice's
// copies are issued before this slice's WMMAs, waited only at iteration end.
// mode 0: write f32 to outF.  mode 1: split-QKV epilogue (q/gate/k/v).
__global__ __launch_bounds__(256) void k_gemm(const _Float16* __restrict__ A,
                                              const _Float16* __restrict__ Bm,
                                              int M, int N, int K, int mode,
                                              float* __restrict__ outF,
                                              _Float16* __restrict__ qO,
                                              float* __restrict__ gateO,
                                              _Float16* __restrict__ kO,
                                              _Float16* __restrict__ vO) {
  __shared__ __align__(16) _Float16 As[2][128][32];   // 16 KB
  __shared__ __align__(16) _Float16 Bs[2][64][32];    //  8 KB
  int tid = threadIdx.x, lane = tid & 31, wave = tid >> 5;
  int m0 = blockIdx.x * 128, n0 = blockIdx.y * 64;
  int wm = (wave & 3) * 32;     // wave M offset in block tile
  int wn = (wave >> 2) * 32;    // wave N offset in block tile
  v8f acc[2][2] = {};

  // per-thread staging coordinates (A: 16 halfs/thread, B: 8 halfs/thread)
  int ar = tid >> 1, ac = (tid & 1) * 16;
  int br = tid >> 2, bc = (tid & 3) * 8;
  const _Float16* Abase = A  + (size_t)(m0 + ar) * K + ac;
  const _Float16* Bbase = Bm + (size_t)(n0 + br) * K + bc;
  int steps = K / 32;

  // prologue: stage slice 0 into buffer 0
  async_copy_b128(Abase,     &As[0][ar][ac]);
  async_copy_b128(Abase + 8, &As[0][ar][ac + 8]);
  async_copy_b128(Bbase,     &Bs[0][br][bc]);
  wait_async();
  __syncthreads();

  for (int i = 0; i < steps; ++i) {
    int cur = i & 1;
    if (i + 1 < steps) {        // prefetch next slice into the other buffer
      const _Float16* An = Abase + (size_t)(i + 1) * 32;
      const _Float16* Bn = Bbase + (size_t)(i + 1) * 32;
      async_copy_b128(An,     &As[cur ^ 1][ar][ac]);
      async_copy_b128(An + 8, &As[cur ^ 1][ar][ac + 8]);
      async_copy_b128(Bn,     &Bs[cur ^ 1][br][bc]);
    }
    v16h a0 = load_frag(&As[cur][wm][0],      32, lane);
    v16h a1 = load_frag(&As[cur][wm + 16][0], 32, lane);
    v16h b0 = load_frag(&Bs[cur][wn][0],      32, lane);
    v16h b1 = load_frag(&Bs[cur][wn + 16][0], 32, lane);
    acc[0][0] = wmma_f16(a0, b0, acc[0][0]);
    acc[0][1] = wmma_f16(a0, b1, acc[0][1]);
    acc[1][0] = wmma_f16(a1, b0, acc[1][0]);
    acc[1][1] = wmma_f16(a1, b1, acc[1][1]);
    wait_async();
    __syncthreads();
  }

  int rb  = (lane >> 4) << 3;   // C-layout: rows r+rb, col lane&15
  int col = lane & 15;
  for (int i = 0; i < 2; ++i)
    for (int j = 0; j < 2; ++j)
      for (int r = 0; r < 8; ++r) {
        int gm = m0 + wm + i * 16 + rb + r;
        int gn = n0 + wn + j * 16 + col;
        float v = acc[i][j][r];
        if (mode == 0) {
          outF[(size_t)gm * N + gn] = v;
        } else {
          if (gn < DIMN)            qO[(size_t)gm * DIMN + gn]                 = (_Float16)v;
          else if (gn < 2 * DIMN)   gateO[(size_t)gm * DIMN + (gn - DIMN)]     = v;
          else if (gn < 2 * DIMN + GQA_DIM)
                                    kO[(size_t)gm * GQA_DIM + (gn - 2 * DIMN)] = (_Float16)v;
          else                      vO[(size_t)gm * GQA_DIM + (gn - 2 * DIMN - GQA_DIM)] = (_Float16)v;
        }
      }
}

// ---------------- kernel 4: RoPE in-place on f16 ----------------
__global__ __launch_bounds__(256) void k_rotary(_Float16* __restrict__ qk,
                                                const float* __restrict__ freqs,
                                                int heads) {
  const int pairs = HEAD_DIM / 2;
  int idx = blockIdx.x * 256 + threadIdx.x;
  int p = idx % pairs;
  int h = (idx / pairs) % heads;
  int t = idx / (pairs * heads);
  if (t >= TTOT) return;
  float f = freqs[(size_t)t * pairs + p];
  float c = cosf(f), s = sinf(f);
  _Float16* ptr = qk + (size_t)t * heads * HEAD_DIM + h * HEAD_DIM + 2 * p;
  float x1 = (float)ptr[0], x2 = (float)ptr[1];
  ptr[0] = (_Float16)(x1 * c - x2 * s);
  ptr[1] = (_Float16)(x1 * s + x2 * c);
}

// ---------------- kernel 5: GQA flash attention + fused gating ----------------
// Block = 8 waves; each wave owns a 16-row Q tile (128 rows / block).
// Loop over 32-key blocks: K staged via async LDS copies, V transposed in LDS,
// S = Q K^T via WMMA, online softmax (16-lane shfl reductions match the
// C-layout half-wave row ownership), P -> A-layout via per-wave LDS, PV WMMA.
__global__ __launch_bounds__(256) void k_flash(const _Float16* __restrict__ Q,
                                               const _Float16* __restrict__ Km,
                                               const _Float16* __restrict__ Vm,
                                               const float* __restrict__ gate,
                                               _Float16* __restrict__ O16) {
  __shared__ __align__(16) _Float16 Ks[32][HEAD_DIM];     // 4 KB
  __shared__ __align__(16) _Float16 Vt[HEAD_DIM][32];     // 4 KB (transposed)
  __shared__ __align__(16) _Float16 Pl[8][16][32];        // 8 KB per-wave P scratch

  int tid = threadIdx.x, lane = tid & 31, wave = tid >> 5;
  int rowTile = blockIdx.x, head = blockIdx.y, batch = blockIdx.z;
  int kvHead = head >> 2;                                 // groups = 4
  int qRow = batch * SEQ + rowTile * 128 + wave * 16;
  const float scale = 0.125f;                             // 1/sqrt(64)

  // Q fragments (16x64 = two 16x32 A-frags), kept in registers
  const _Float16* qbase = Q + (size_t)qRow * DIMN + head * HEAD_DIM;
  v16h qa0 = load_frag(qbase,      DIMN, lane);
  v16h qa1 = load_frag(qbase + 32, DIMN, lane);

  v8f  oacc[4] = {};
  float row_max[8], row_sum[8];
  for (int r = 0; r < 8; ++r) { row_max[r] = -1e30f; row_sum[r] = 0.f; }

  for (int kb = 0; kb < SEQ / 32; ++kb) {
    int keyBase = kb * 32;
    { // stage K block (async -> LDS, row-major) and V block (transposed)
      int r = tid >> 3, c = (tid & 7) * 8;
      const _Float16* gk = Km + (size_t)(batch * SEQ + keyBase + r) * GQA_DIM + kvHead * HEAD_DIM + c;
      async_copy_b128(gk, &Ks[r][c]);
      const _Float16* gv = Vm + (size_t)(batch * SEQ + keyBase + r) * GQA_DIM + kvHead * HEAD_DIM + c;
      v8h vv = *(const v8h*)gv;
      for (int i = 0; i < 8; ++i) Vt[c + i][r] = vv[i];
    }
    wait_async();
    __syncthreads();

    // S tiles: 16 rows x 32 keys (two 16x16 n-tiles, 2 K-steps over d)
    v8f s0 = {}, s1 = {};
    s0 = wmma_f16(qa0, load_frag(&Ks[0][0],   HEAD_DIM, lane), s0);
    s0 = wmma_f16(qa1, load_frag(&Ks[0][32],  HEAD_DIM, lane), s0);
    s1 = wmma_f16(qa0, load_frag(&Ks[16][0],  HEAD_DIM, lane), s1);
    s1 = wmma_f16(qa1, load_frag(&Ks[16][32], HEAD_DIM, lane), s1);

    // online softmax
    for (int r = 0; r < 8; ++r) {
      float a0 = s0[r] * scale, a1 = s1[r] * scale;
      float mx = fmaxf(a0, a1);
      for (int m = 8; m >= 1; m >>= 1) mx = fmaxf(mx, __shfl_xor(mx, m, 32));
      float mnew = fmaxf(row_max[r], mx);
      float corr = __expf(row_max[r] - mnew);
      row_max[r] = mnew;
      float p0 = __expf(a0 - mnew), p1 = __expf(a1 - mnew);
      float ps = p0 + p1;
      for (int m = 8; m >= 1; m >>= 1) ps += __shfl_xor(ps, m, 32);
      row_sum[r] = row_sum[r] * corr + ps;
      oacc[0][r] *= corr; oacc[1][r] *= corr;
      oacc[2][r] *= corr; oacc[3][r] *= corr;
      int prow = r + ((lane >> 4) << 3);
      Pl[wave][prow][lane & 15]        = (_Float16)p0;
      Pl[wave][prow][(lane & 15) + 16] = (_Float16)p1;
    }

    // PV: P (16x32) x V (32x64); Vt gives contiguous B-frag reads
    v16h pa = load_frag(&Pl[wave][0][0], 32, lane);
    for (int t = 0; t < 4; ++t)
      oacc[t] = wmma_f16(pa, load_frag(&Vt[t * 16][0], 32, lane), oacc[t]);

    __syncthreads();
  }

  // epilogue: normalize, fuse sigmoid gating, store f16
  int rb  = (lane >> 4) << 3;
  int col = lane & 15;
  for (int t = 0; t < 4; ++t)
    for (int r = 0; r < 8; ++r) {
      int row = qRow + rb + r;
      int cAbs = head * HEAD_DIM + t * 16 + col;
      float ov = oacc[t][r] / row_sum[r];
      float gv = gate[(size_t)row * DIMN + cAbs];
      float sg = 1.f / (1.f + __expf(-gv));
      O16[(size_t)row * DIMN + cAbs] = (_Float16)(ov * sg);
    }
}

// ---------------- host launcher ----------------
extern "C" void kernel_launch(void* const* d_in, const int* in_sizes, int n_in,
                              void* d_out, int out_size, void* d_ws, size_t ws_size,
                              hipStream_t stream) {
  const float* x     = (const float*)d_in[0];
  const float* freqs = (const float*)d_in[1];
  const float* g     = (const float*)d_in[2];
  const float* W_qkv = (const float*)d_in[3];
  const float* W_out = (const float*)d_in[4];
  float* out = (float*)d_out;

  size_t off = 0;
  auto alloc = [&](size_t bytes) -> void* {
    void* p = (char*)d_ws + off;
    off += (bytes + 255) & ~(size_t)255;
    return p;
  };
  _Float16* hN     = (_Float16*)alloc((size_t)TTOT * DIMN * 2);
  _Float16* Wqkv16 = (_Float16*)alloc((size_t)QKV_N * DIMN * 2);
  _Float16* Wout16 = (_Float16*)alloc((size_t)DIMN * DIMN * 2);
  _Float16* q16    = (_Float16*)alloc((size_t)TTOT * DIMN * 2);
  _Float16* k16    = (_Float16*)alloc((size_t)TTOT * GQA_DIM * 2);
  _Float16* v16b   = (_Float16*)alloc((size_t)TTOT * GQA_DIM * 2);
  float*    gateB  = (float*)   alloc((size_t)TTOT * DIMN * 4);
  _Float16* o16    = (_Float16*)alloc((size_t)TTOT * DIMN * 2);

  // 1) RMSNorm + cast
  k_rmsnorm<<<TTOT, 256, 0, stream>>>(x, g, hN);
  // 2) weight casts
  k_cvt16<<<(QKV_N * DIMN) / 256, 256, 0, stream>>>(W_qkv, Wqkv16, QKV_N * DIMN);
  k_cvt16<<<(DIMN * DIMN) / 256, 256, 0, stream>>>(W_out, Wout16, DIMN * DIMN);
  // 3) QKV GEMM with split epilogue
  k_gemm<<<dim3(TTOT / 128, QKV_N / 64), 256, 0, stream>>>(
      hN, Wqkv16, TTOT, QKV_N, DIMN, /*mode=*/1,
      nullptr, q16, gateB, k16, v16b);
  // 4) RoPE
  k_rotary<<<(TTOT * Q_HEADS * (HEAD_DIM / 2)) / 256, 256, 0, stream>>>(q16, freqs, Q_HEADS);
  k_rotary<<<(TTOT * KV_HEADS * (HEAD_DIM / 2)) / 256, 256, 0, stream>>>(k16, freqs, KV_HEADS);
  // 5) flash attention + fused sigmoid gating
  k_flash<<<dim3(SEQ / 128, Q_HEADS, BATCH), 256, 0, stream>>>(q16, k16, v16b, gateB, o16);
  // 6) out-projection
  k_gemm<<<dim3(TTOT / 128, DIMN / 64), 256, 0, stream>>>(
      o16, Wout16, TTOT, DIMN, DIMN, /*mode=*/0,
      out, nullptr, nullptr, nullptr, nullptr);
}
